// TopK_RNN_T_Decoder_4698694222300
// MI455X (gfx1250) — matrix operand order repaired and checked by reading, not compiled, not run on hardware
//
#include <hip/hip_runtime.h>

// ---------------------------------------------------------------------------
// Types & helpers
// ---------------------------------------------------------------------------
typedef unsigned short u16;
typedef unsigned int   u32;
typedef __attribute__((ext_vector_type(16))) __bf16 v16bf;
typedef __attribute__((ext_vector_type(8)))  float  v8f;

__device__ __forceinline__ u16 f2bf(float x) {
    u32 u = __float_as_uint(x);
    u32 r = u + 0x7FFFu + ((u >> 16) & 1u);   // round-to-nearest-even
    return (u16)(r >> 16);
}
__device__ __forceinline__ float bf2f(u16 h) {
    return __uint_as_float(((u32)h) << 16);
}
__device__ __forceinline__ float sigf(float x) {
    return 1.0f / (1.0f + __expf(-x));
}

__device__ __forceinline__ v16bf ld_frag(const u16* p0, const u16* p1) {
    union { v16bf v; uint4 q[2]; } u;
    u.q[0] = *reinterpret_cast<const uint4*>(p0);
    u.q[1] = *reinterpret_cast<const uint4*>(p1);
    return u.v;
}
// A fragment (16x32 bf16): lane holds row m = lane%16; K = half*8..+8 and 16+half*8..+8
__device__ __forceinline__ v16bf load_a(const u16* base, size_t lda, int m, int half, int k0) {
    const u16* p = base + (size_t)m * lda + (size_t)(k0 + half * 8);
    return ld_frag(p, p + 16);
}
// B fragment (32x16 bf16): lane holds col n; K = half*8..+8 and 16+half*8..+8, from row-major W[n][k]
__device__ __forceinline__ v16bf load_b(const u16* base, size_t ldb, int n, int half, int k0) {
    const u16* p = base + (size_t)n * ldb + (size_t)(k0 + half * 8);
    return ld_frag(p, p + 16);
}
__device__ __forceinline__ v8f wmma_bf16(v16bf a, v16bf b, v8f c) {
    return __builtin_amdgcn_wmma_f32_16x16x32_bf16(false, a, false, b, (short)0, c, false, false);
}

// ---------------------------------------------------------------------------
// Problem constants
// ---------------------------------------------------------------------------
static constexpr int BB = 16, TT = 1024, FF = 80, FP = 96;
static constexpr int HH = 1024, PP = 512, EE = 256, HD = 512, JJ = 512;
static constexpr int VV = 54, MAXLEN = 55, WBEAM = 4;

// Workspace layout (bytes, 256-aligned)
static constexpr size_t OFF_BAR   = 0;                                  // 256 B barrier
static constexpr size_t OFF_XPAD  = 256;                                // 16384*96*2
static constexpr size_t OFF_WIH   = OFF_XPAD  + (size_t)BB*TT*FP*2;     // 4096*96*2
static constexpr size_t OFF_WHH   = OFF_WIH   + (size_t)4*HH*FP*2;      // 4096*1024*2
static constexpr size_t OFF_PROJW = OFF_WHH   + (size_t)4*HH*HH*2;      // 512*1024*2
static constexpr size_t OFF_EMB   = OFF_PROJW + (size_t)PP*HH*2;        // 54*256*2 (pad)
static constexpr size_t OFF_DWIH  = OFF_EMB   + 32768;                  // 2048*256*2
static constexpr size_t OFF_DWHH  = OFF_DWIH  + (size_t)4*HD*EE*2;      // 2048*512*2
static constexpr size_t OFF_WE    = OFF_DWHH  + (size_t)4*HD*HD*2;      // 512*512*2
static constexpr size_t OFF_WD    = OFF_WE    + (size_t)JJ*PP*2;        // 512*512*2
static constexpr size_t OFF_HALL  = OFF_WD    + (size_t)JJ*HD*2;        // 16384*1024*2
static constexpr size_t OFF_EPROJ = OFF_HALL  + (size_t)BB*TT*HH*2;     // 16384*512*2
static constexpr size_t OFF_EALL  = OFF_EPROJ + (size_t)BB*TT*PP*2;     // 16384*512*4 (f32)

// ---------------------------------------------------------------------------
// f32 -> bf16 conversion with column zero-padding
// ---------------------------------------------------------------------------
__global__ void conv_bf16_pad(const float* __restrict__ src, u16* __restrict__ dst,
                              int rows, int cols, int colsPad) {
    size_t total = (size_t)rows * colsPad;
    for (size_t i = (size_t)blockIdx.x * blockDim.x + threadIdx.x; i < total;
         i += (size_t)gridDim.x * blockDim.x) {
        int r = (int)(i / colsPad), c = (int)(i % colsPad);
        float v = (c < cols) ? src[(size_t)r * cols + c] : 0.0f;
        dst[i] = f2bf(v);
    }
}

// ---------------------------------------------------------------------------
// Generic WMMA GEMM: C[M x N] = A_bf16[M x K] @ B_bf16[N x K]^T + bias
// Register-blocked: each wave computes 4 N-tiles, reusing one A fragment per
// K-chunk across 4 WMMAs (4x less A traffic from L2).
// grid.x = N/512 (8 waves x 4 tiles = 512 cols per block), grid.y = M/16
// ---------------------------------------------------------------------------
__global__ __launch_bounds__(256, 1) void gemm_bf16(
    const u16* __restrict__ A, size_t lda,
    const u16* __restrict__ B, size_t ldb,
    const float* __restrict__ bias,
    u16* __restrict__ Cbf, float* __restrict__ Cf, size_t ldc, int K) {
    const int wave = threadIdx.x >> 5, lane = threadIdx.x & 31;
    const int half = lane >> 4, l16 = lane & 15;
    const int nbase0 = (blockIdx.x * 32 + wave * 4) * 16;   // first of 4 n-tiles
    const int mbase  = blockIdx.y * 16;
    v8f acc[4];
#pragma unroll
    for (int q = 0; q < 4; ++q) {
        float bv = bias ? bias[nbase0 + q * 16 + l16] : 0.0f;
#pragma unroll
        for (int r = 0; r < 8; ++r) acc[q][r] = bv;
    }
    const u16* Abase = A + (size_t)mbase * lda;
    for (int k0 = 0; k0 < K; k0 += 32) {
        v16bf a = load_a(Abase, lda, l16, half, k0);
#pragma unroll
        for (int q = 0; q < 4; ++q)
            acc[q] = wmma_bf16(a, load_b(B, ldb, nbase0 + q * 16 + l16, half, k0), acc[q]);
    }
#pragma unroll
    for (int q = 0; q < 4; ++q) {
#pragma unroll
        for (int r = 0; r < 8; ++r) {
            int m = mbase + r + 8 * half;
            int n = nbase0 + q * 16 + l16;
            if (Cf)  Cf [(size_t)m * ldc + n] = acc[q][r];
            if (Cbf) Cbf[(size_t)m * ldc + n] = f2bf(acc[q][r]);
        }
    }
}

// ---------------------------------------------------------------------------
// Persistent encoder LSTM: 16 blocks x 128 threads = 64 waves, one 16-col
// h-tile per wave; c-state lives in VGPRs; grid barrier between timesteps.
// ---------------------------------------------------------------------------
__global__ __launch_bounds__(128, 1) void enc_lstm_kernel(
    const u16* __restrict__ xpad,  // [B][T][96] bf16
    const u16* __restrict__ Wih,   // [4H][96]   bf16
    const u16* __restrict__ Whh,   // [4H][H]    bf16
    const float* __restrict__ bias,
    u16* __restrict__ hall,        // [B][T][H]  bf16 out
    u32* __restrict__ barcnt) {
    const int wave = threadIdx.x >> 5, lane = threadIdx.x & 31;
    const int half = lane >> 4, l16 = lane & 15;
    const int ct  = blockIdx.x * 4 + wave;  // 0..63
    const int col = ct * 16 + l16;          // this lane's h column

    float bv[4];
#pragma unroll
    for (int g = 0; g < 4; ++g) bv[g] = bias[g * HH + col];
    float cst[8];
#pragma unroll
    for (int r = 0; r < 8; ++r) cst[r] = 0.0f;

    for (int t = 0; t < TT; ++t) {
        // Warm WGP$ for the next timestep's x row while we compute this one.
        if (t + 1 < TT)
            __builtin_prefetch(xpad + (size_t)(t + 1) * FP + (size_t)l16 * (TT * FP), 0, 1);

        v8f acc[4];
#pragma unroll
        for (int g = 0; g < 4; ++g)
#pragma unroll
            for (int r = 0; r < 8; ++r) acc[g][r] = bv[g];

        // x @ Wih^T   (K = 96 padded)
        const u16* xa = xpad + (size_t)t * FP;
        for (int k0 = 0; k0 < FP; k0 += 32) {
            v16bf a = load_a(xa, (size_t)TT * FP, l16, half, k0);
#pragma unroll
            for (int g = 0; g < 4; ++g)
                acc[g] = wmma_bf16(a, load_b(Wih, FP, g * HH + col, half, k0), acc[g]);
        }
        // h_{t-1} @ Whh^T  (K = 1024)
        if (t > 0) {
            const u16* ha = hall + (size_t)(t - 1) * HH;
            for (int k0 = 0; k0 < HH; k0 += 32) {
                v16bf a = load_a(ha, (size_t)TT * HH, l16, half, k0);
#pragma unroll
                for (int g = 0; g < 4; ++g)
                    acc[g] = wmma_bf16(a, load_b(Whh, HH, g * HH + col, half, k0), acc[g]);
            }
        }
        // LSTM cell, write h_t
#pragma unroll
        for (int r = 0; r < 8; ++r) {
            int m = r + 8 * half;
            float cn = sigf(acc[1][r]) * cst[r] + sigf(acc[0][r]) * tanhf(acc[2][r]);
            cst[r] = cn;
            float hn = sigf(acc[3][r]) * tanhf(cn);
            hall[(size_t)m * (TT * HH) + (size_t)t * HH + col] = f2bf(hn);
        }
        // grid barrier (monotonic counter; zeroed by memset before launch)
        __syncthreads();
        if (threadIdx.x == 0) {
            __threadfence();
            __hip_atomic_fetch_add(barcnt, 1u, __ATOMIC_ACQ_REL, __HIP_MEMORY_SCOPE_AGENT);
            u32 target = (u32)(t + 1) * 16u;
            while (__hip_atomic_load(barcnt, __ATOMIC_ACQUIRE, __HIP_MEMORY_SCOPE_AGENT) < target)
                __builtin_amdgcn_s_sleep(2);
        }
        __syncthreads();
    }
}

// ---------------------------------------------------------------------------
// Beam search: one 256-thread workgroup per batch element. Beams padded to
// M=16 for WMMA (rows 4..15 zero). State in LDS.
// ---------------------------------------------------------------------------
__global__ __launch_bounds__(256, 1) void beam_kernel(
    const float* __restrict__ eall,   // [B][T][J] f32 : enc@We^T + bj
    const u16* __restrict__ embB,     // [V][E] bf16
    const u16* __restrict__ Wd,       // [J][Hd] bf16
    const u16* __restrict__ dWih,     // [4Hd][E] bf16
    const u16* __restrict__ dWhh,     // [4Hd][Hd] bf16
    const float* __restrict__ db,     // [4Hd]
    const float* __restrict__ Wo,     // [V][J] f32
    const float* __restrict__ bo,     // [V]
    const int* __restrict__ lengths,  // [B]
    int* __restrict__ out) {
    const int tid  = threadIdx.x;
    const int wave = tid >> 5, lane = tid & 31;
    const int half = lane >> 4, l16 = lane & 15;
    const int b    = blockIdx.x;

    __shared__ u16   hb[16 * HD];     // beam h state, bf16, rows 4..15 zero
    __shared__ u16   xb[16 * EE];     // embed(tok) staging, rows 4..15 zero
    __shared__ u16   hp[WBEAM * HD];  // gathered parent h (bf16)
    __shared__ u16   htmp[WBEAM * HD];
    __shared__ float cf[WBEAM * HD];  // c state (f32)
    __shared__ float cp[WBEAM * HD];  // gathered parent c
    __shared__ u16   jb[WBEAM * JJ];  // tanh(e + d), bf16
    __shared__ float lg[WBEAM * VV];
    __shared__ float cand[WBEAM * VV];
    __shared__ int   seqbuf[2][WBEAM][56];
    __shared__ int   lensbuf[2][WBEAM];
    __shared__ float cum[WBEAM], vals[WBEAM], lsm[WBEAM];
    __shared__ int   parent[WBEAM], tokv[WBEAM], emitf[WBEAM];

    // init
    for (int i = tid; i < 16 * HD; i += 256) hb[i] = 0;
    for (int i = tid; i < 16 * EE; i += 256) xb[i] = 0;
    for (int i = tid; i < WBEAM * HD; i += 256) { cf[i] = 0.f; cp[i] = 0.f; hp[i] = 0; htmp[i] = 0; }
    if (tid < WBEAM) {
        cum[tid] = (tid == 0) ? 0.0f : -1.0e9f;
        lensbuf[0][tid] = 0; lensbuf[1][tid] = 0;
        for (int i = 0; i < 56; ++i) { seqbuf[0][tid][i] = 0; seqbuf[1][tid][i] = 0; }
    }
    __syncthreads();

    int len = lengths[b];
    if (len > TT) len = TT;
    if (len < 0) len = 0;
    int cur = 0;

    for (int t = 0; t < len; ++t) {
        const int nxt = cur ^ 1;
        // --- Phase A: d = hb @ Wd^T ; jb = tanh(e_t + d) ---------------------
        {
            v8f acc[4];
#pragma unroll
            for (int q = 0; q < 4; ++q)
#pragma unroll
                for (int r = 0; r < 8; ++r) acc[q][r] = 0.0f;
            for (int k0 = 0; k0 < HD; k0 += 32) {
                v16bf a = load_a(hb, HD, l16, half, k0);
#pragma unroll
                for (int q = 0; q < 4; ++q)
                    acc[q] = wmma_bf16(a, load_b(Wd, HD, (wave * 4 + q) * 16 + l16, half, k0),
                                       acc[q]);
            }
            if (half == 0) {
#pragma unroll
                for (int q = 0; q < 4; ++q) {
                    int n = (wave * 4 + q) * 16 + l16;
                    float e = eall[((size_t)b * TT + t) * JJ + n];
#pragma unroll
                    for (int r = 0; r < 4; ++r)
                        jb[r * JJ + n] = f2bf(tanhf(e + acc[q][r]));
                }
            }
        }
        __syncthreads();
        // --- Phase B: logits = jb @ Wo^T + bo (VALU, V=54 tiny) --------------
        if (tid < WBEAM * VV) {
            int m = tid / VV, v = tid % VV;
            float s = bo[v];
            const float* wr = Wo + (size_t)v * JJ;
            for (int k = 0; k < JJ; ++k) s += bf2f(jb[m * JJ + k]) * wr[k];
            lg[tid] = s;
        }
        __syncthreads();
        // --- Phase C: log-softmax + candidates -------------------------------
        if (tid < WBEAM) {
            float mx = -3.0e38f;
            for (int v = 0; v < VV; ++v) mx = fmaxf(mx, lg[tid * VV + v]);
            float sm = 0.0f;
            for (int v = 0; v < VV; ++v) sm += __expf(lg[tid * VV + v] - mx);
            lsm[tid] = mx + logf(sm);
        }
        __syncthreads();
        if (tid < WBEAM * VV) {
            int m = tid / VV;
            cand[tid] = cum[m] + (lg[tid] - lsm[m]);
        }
        __syncthreads();
        // --- Phase D: top-4 --------------------------------------------------
        if (tid == 0) {
            for (int w = 0; w < WBEAM; ++w) {
                float best = -3.0e38f; int bi = 0;
                for (int i = 0; i < WBEAM * VV; ++i)
                    if (cand[i] > best) { best = cand[i]; bi = i; }
                parent[w] = bi / VV; tokv[w] = bi % VV; vals[w] = best;
                cand[bi] = -3.0e38f;
            }
        }
        __syncthreads();
        // --- Phase E: sequence/len/cum update + gathers ----------------------
        if (tid < WBEAM) {
            int p = parent[tid], tk = tokv[tid];
            int Lold = lensbuf[cur][p];
            int em = (tk != 0) && (Lold < MAXLEN - 1);
            emitf[tid] = em;
            for (int i = 0; i < 56; ++i) seqbuf[nxt][tid][i] = seqbuf[cur][p][i];
            if (em) seqbuf[nxt][tid][Lold] = tk;
            lensbuf[nxt][tid] = Lold + em;
            cum[tid] = vals[tid];
        }
        for (int i = tid; i < WBEAM * HD; i += 256) {
            int m = i >> 9, k = i & (HD - 1);
            hp[i] = hb[parent[m] * HD + k];
            cp[i] = cf[parent[m] * HD + k];
        }
        for (int i = tid; i < WBEAM * EE; i += 256) {
            int m = i >> 8, k = i & (EE - 1);
            xb[m * EE + k] = embB[(size_t)tokv[m] * EE + k];
        }
        __syncthreads();
        for (int i = tid; i < WBEAM * HD; i += 256) hb[i] = hp[i];  // parent h -> A operand
        __syncthreads();
        // --- Phase F: decoder LSTM cell via WMMA -----------------------------
        for (int q = 0; q < 4; ++q) {
            int colb = (wave * 4 + q) * 16;
            v8f acc[4];
#pragma unroll
            for (int g = 0; g < 4; ++g) {
                float bvv = db[g * HD + colb + l16];
#pragma unroll
                for (int r = 0; r < 8; ++r) acc[g][r] = bvv;
            }
            for (int k0 = 0; k0 < EE; k0 += 32) {       // embed part (K=256)
                v16bf a = load_a(xb, EE, l16, half, k0);
#pragma unroll
                for (int g = 0; g < 4; ++g)
                    acc[g] = wmma_bf16(a, load_b(dWih, EE, g * HD + colb + l16, half, k0), acc[g]);
            }
            for (int k0 = 0; k0 < HD; k0 += 32) {       // hidden part (K=512)
                v16bf a = load_a(hb, HD, l16, half, k0);
#pragma unroll
                for (int g = 0; g < 4; ++g)
                    acc[g] = wmma_bf16(a, load_b(dWhh, HD, g * HD + colb + l16, half, k0), acc[g]);
            }
            if (half == 0) {
#pragma unroll
                for (int r = 0; r < 4; ++r) {
                    int m = r, col = colb + l16;
                    float cold = cp[m * HD + col];
                    float cn = sigf(acc[1][r]) * cold + sigf(acc[0][r]) * tanhf(acc[2][r]);
                    float hn = sigf(acc[3][r]) * tanhf(cn);
                    int em = emitf[m];
                    cf[m * HD + col]   = em ? cn : cold;
                    htmp[m * HD + col] = em ? f2bf(hn) : hp[m * HD + col];
                }
            }
        }
        __syncthreads();
        for (int i = tid; i < WBEAM * HD; i += 256) hb[i] = htmp[i];
        cur = nxt;
        __syncthreads();
    }

    // finalize: EOS + write outputs
    if (tid < WBEAM) seqbuf[cur][tid][lensbuf[cur][tid]] = 53;  // EOS
    __syncthreads();
    for (int i = tid; i < WBEAM * MAXLEN; i += 256) {
        int m = i / MAXLEN, k = i % MAXLEN;
        out[((size_t)b * WBEAM + m) * MAXLEN + k] = seqbuf[cur][m][k];
    }
    for (int i = tid; i < MAXLEN; i += 256)
        out[(size_t)BB * WBEAM * MAXLEN + (size_t)b * MAXLEN + i] = seqbuf[cur][0][i];
}

// ---------------------------------------------------------------------------
// Host entry
// ---------------------------------------------------------------------------
extern "C" void kernel_launch(void* const* d_in, const int* in_sizes, int n_in,
                              void* d_out, int out_size, void* d_ws, size_t ws_size,
                              hipStream_t stream) {
    (void)in_sizes; (void)n_in; (void)out_size; (void)ws_size;
    char* ws = (char*)d_ws;

    const float* inputs  = (const float*)d_in[0];
    const int*   lens    = (const int*)d_in[1];
    const float* encWih  = (const float*)d_in[3];
    const float* encWhh  = (const float*)d_in[4];
    const float* encB    = (const float*)d_in[5];
    const float* projW   = (const float*)d_in[6];
    const float* projB   = (const float*)d_in[7];
    const float* embed   = (const float*)d_in[8];
    const float* decWih  = (const float*)d_in[9];
    const float* decWhh  = (const float*)d_in[10];
    const float* decB    = (const float*)d_in[11];
    const float* jWe     = (const float*)d_in[12];
    const float* jWd     = (const float*)d_in[13];
    const float* jBj     = (const float*)d_in[14];
    const float* jWo     = (const float*)d_in[15];
    const float* jBo     = (const float*)d_in[16];

    u16* xpad   = (u16*)(ws + OFF_XPAD);
    u16* wihB   = (u16*)(ws + OFF_WIH);
    u16* whhB   = (u16*)(ws + OFF_WHH);
    u16* projWB = (u16*)(ws + OFF_PROJW);
    u16* embB   = (u16*)(ws + OFF_EMB);
    u16* dWihB  = (u16*)(ws + OFF_DWIH);
    u16* dWhhB  = (u16*)(ws + OFF_DWHH);
    u16* weB    = (u16*)(ws + OFF_WE);
    u16* wdB    = (u16*)(ws + OFF_WD);
    u16* hall   = (u16*)(ws + OFF_HALL);
    u16* eproj  = (u16*)(ws + OFF_EPROJ);
    float* eall = (float*)(ws + OFF_EALL);
    u32* barcnt = (u32*)(ws + OFF_BAR);

    hipMemsetAsync(ws + OFF_BAR, 0, 256, stream);

    auto conv = [&](const float* s, u16* d, int r, int c, int cp) {
        size_t n = (size_t)r * cp;
        int blocks = (int)((n + 255) / 256);
        if (blocks > 16384) blocks = 16384;
        conv_bf16_pad<<<blocks, 256, 0, stream>>>(s, d, r, c, cp);
    };
    conv(inputs, xpad,   BB * TT, FF, FP);
    conv(encWih, wihB,   4 * HH,  FF, FP);
    conv(encWhh, whhB,   4 * HH,  HH, HH);
    conv(projW,  projWB, PP,      HH, HH);
    conv(embed,  embB,   VV,      EE, EE);
    conv(decWih, dWihB,  4 * HD,  EE, EE);
    conv(decWhh, dWhhB,  4 * HD,  HD, HD);
    conv(jWe,    weB,    JJ,      PP, PP);
    conv(jWd,    wdB,    JJ,      HD, HD);

    // Encoder LSTM (persistent, grid barrier per timestep)
    enc_lstm_kernel<<<16, 128, 0, stream>>>(xpad, wihB, whhB, encB, hall, barcnt);

    // enc_proj = h_all @ projW^T + proj_b  (M=16384, N=512, K=1024) -> bf16
    gemm_bf16<<<dim3(PP / 512, (BB * TT) / 16), 256, 0, stream>>>(
        hall, HH, projWB, HH, projB, eproj, nullptr, PP, HH);

    // e_all = enc_proj @ We^T + bj  (M=16384, N=512, K=512) -> f32
    gemm_bf16<<<dim3(JJ / 512, (BB * TT) / 16), 256, 0, stream>>>(
        eproj, PP, weB, PP, jBj, nullptr, eall, JJ, PP);

    // Beam search, one workgroup per batch element
    beam_kernel<<<BB, 256, 0, stream>>>(eall, embB, wdB, dWihB, dWhhB,
                                        decB, jWo, jBo, lens, (int*)d_out);
}
